// HandWritingSynthesisNet_37933151158379
// MI455X (gfx1250) — matrix operand order repaired and checked by reading, not compiled, not run on hardware
//
#include <hip/hip_runtime.h>
#include <stddef.h>

// ---------------------------------------------------------------------------
// Handwriting synthesis net (Graves attention LSTM) for MI455X / gfx1250.
// bf16 WMMA (v_wmma_f32_16x16x32_bf16) everywhere, fp32 accumulate.
// Per-step A matrices live in the 320KB WGP LDS; global->LDS staging uses
// GLOBAL_LOAD_ASYNC_TO_LDS_B128 (ASYNCcnt) when the toolchain exposes it.
// ---------------------------------------------------------------------------

#define B_   128
#define T_   600
#define U_   64
#define H_   400
#define V_   77
#define KK_  10
#define O_   121
#define G_   1600           // 4*H gates
#define KL1  480            // [h1(400), w(77), x(3)]
#define KL23 896            // [hprev(400), wv(77), x(3), hself(400), pad(16)]
#define KATT 416            // attention K (h 400 + 16 cols hitting Wwp zeros)
#define KOUT 1216           // hcat padded 1200->1216
#define WVS  80             // WV row stride (16B aligned)
#define KCH  64             // k-chunk for output GEMM LDS staging

typedef __attribute__((ext_vector_type(16))) __bf16        v16bf;
typedef __attribute__((ext_vector_type(8)))  float         v8f;
typedef __attribute__((ext_vector_type(8)))  unsigned int  v8u;
typedef __attribute__((ext_vector_type(4)))  unsigned int  v4u;
typedef __attribute__((ext_vector_type(4)))  int           v4i;

#if defined(__has_builtin)
#if __has_builtin(__builtin_amdgcn_global_load_async_to_lds_b128)
#define ASYNC_EN 1
#endif
#endif
#ifndef ASYNC_EN
#define ASYNC_EN 0
#endif

typedef __attribute__((address_space(1))) v4i* gas1_v4i;   // global int4*
typedef __attribute__((address_space(3))) v4i* las3_v4i;   // LDS int4*

// 16-byte global -> LDS copy (async when available).
__device__ __forceinline__ void cp16_g2l(const unsigned short* __restrict__ g,
                                         unsigned short* __restrict__ l) {
#if ASYNC_EN
    __builtin_amdgcn_global_load_async_to_lds_b128((gas1_v4i)(g), (las3_v4i)(l), 0, 0);
#else
    *(v4u*)l = *(const v4u*)g;
#endif
}
__device__ __forceinline__ void async_wait0() {
#if ASYNC_EN
#if defined(__has_builtin) && __has_builtin(__builtin_amdgcn_s_wait_asynccnt)
    __builtin_amdgcn_s_wait_asynccnt(0);
#else
    asm volatile("s_wait_asynccnt 0" ::: "memory");
#endif
#endif
}

__device__ __forceinline__ unsigned short f2bf(float f) {
    unsigned int u = __float_as_uint(f);
    u = (u + 0x7FFFu + ((u >> 16) & 1u)) >> 16;   // round-to-nearest-even
    return (unsigned short)u;
}
__device__ __forceinline__ float sigm(float x) { return 1.0f / (1.0f + __expf(-x)); }

// One 16x16 tile, K = 32*kt_count. arow/brow pre-offset to the lane's row.
// A: [M,K] row-major bf16 (LDS or global), W: [N,K] row-major bf16 (B = W^T).
__device__ __forceinline__ v8f wmma_tile_k(const unsigned short* __restrict__ arow,
                                           const unsigned short* __restrict__ brow,
                                           int kt_count, int half, v8f acc) {
    for (int kt = 0; kt < kt_count; ++kt) {
        const int k0 = kt * 32;
        v8u au, bu;
#pragma unroll
        for (int v = 0; v < 8; ++v) {
            const int ka = k0 + 2 * v + 16 * half;
            au[v] = *(const unsigned int*)(arow + ka);
            const int kb = k0 + 2 * v + 16 * half;
            bu[v] = *(const unsigned int*)(brow + kb);
        }
        acc = __builtin_amdgcn_wmma_f32_16x16x32_bf16(
                  false, __builtin_bit_cast(v16bf, au),
                  false, __builtin_bit_cast(v16bf, bu),
                  (short)0, acc, false, false);
    }
    return acc;
}

// Gates GEMM: z[128,1600] = A[128,KA] @ W[1600,KA]^T + bias; A resides in LDS.
__device__ __forceinline__ void gates_gemm(const unsigned short* __restrict__ A, int lda,
                                           const unsigned short* __restrict__ W, int ldw,
                                           int kt_count,
                                           const float* __restrict__ bias,
                                           float* __restrict__ zbuf,
                                           int wave, int lane) {
    const int half = lane >> 4, q = lane & 15;
    for (int tile = wave; tile < 800; tile += 32) {
        const int mt = tile & 7, nt = tile >> 3;
        const int m0 = mt * 16, n0 = nt * 16;
        if (tile + 32 < 800)    // prefetch next tile's weight row into caches
            __builtin_prefetch(W + (size_t)(((tile + 32) >> 3) * 16 + q) * ldw, 0, 1);
        v8f acc = {};
        acc = wmma_tile_k(A + (size_t)(m0 + q) * lda,
                          W + (size_t)(n0 + q) * ldw, kt_count, half, acc);
        const float bb = bias[n0 + q];
#pragma unroll
        for (int i = 0; i < 8; ++i) {                 // D: m = i + 8*half, n = q
            const int m = m0 + i + 8 * half;
            zbuf[(size_t)m * G_ + n0 + q] = acc[i] + bb;
        }
    }
}

// --------------------------- weight packing --------------------------------
// layer1: W1cat[n,k], k<400 -> Whh1[n,k]; k<477 -> Wih1[n,3+(k-400)] (w part);
//         else Wih1[n,k-477] (x part).
__global__ void k_cvt1(const float* __restrict__ Wih, const float* __restrict__ Whh,
                       unsigned short* __restrict__ dst) {
    for (int i = blockIdx.x * blockDim.x + threadIdx.x; i < G_ * KL1;
         i += blockDim.x * gridDim.x) {
        const int n = i / KL1, k = i % KL1;
        float v;
        if (k < 400)      v = Whh[(size_t)n * 400 + k];
        else if (k < 477) v = Wih[(size_t)n * 80 + 3 + (k - 400)];
        else              v = Wih[(size_t)n * 80 + (k - 477)];
        dst[i] = f2bf(v);
    }
}
// layers 2/3: [hprev | wv | x | hself | pad], Wih cols: 0..2 x, 3..402 h, 403..479 wv.
__global__ void k_cvt23(const float* __restrict__ Wih, const float* __restrict__ Whh,
                        unsigned short* __restrict__ dst) {
    for (int i = blockIdx.x * blockDim.x + threadIdx.x; i < G_ * KL23;
         i += blockDim.x * gridDim.x) {
        const int n = i / KL23, k = i % KL23;
        float v = 0.0f;
        if (k < 400)      v = Wih[(size_t)n * 480 + 3 + k];
        else if (k < 477) v = Wih[(size_t)n * 480 + 403 + (k - 400)];
        else if (k < 480) v = Wih[(size_t)n * 480 + (k - 477)];
        else if (k < 880) v = Whh[(size_t)n * 400 + (k - 480)];
        dst[i] = f2bf(v);
    }
}
// zero-padded copy: dst[N,K] <- src[Nsrc,Ksrc]
__global__ void k_cvt_pad(const float* __restrict__ src, unsigned short* __restrict__ dst,
                          int N, int K, int Nsrc, int Ksrc) {
    for (int i = blockIdx.x * blockDim.x + threadIdx.x; i < N * K;
         i += blockDim.x * gridDim.x) {
        const int n = i / K, k = i % K;
        dst[i] = f2bf((n < Nsrc && k < Ksrc) ? src[(size_t)n * Ksrc + k] : 0.0f);
    }
}
__global__ void k_pad_hcat(unsigned short* __restrict__ HCAT) {
    const size_t total = (size_t)T_ * B_ * 16;
    for (size_t i = (size_t)blockIdx.x * blockDim.x + threadIdx.x; i < total;
         i += (size_t)blockDim.x * gridDim.x)
        HCAT[(i >> 4) * KOUT + 1200 + (int)(i & 15)] = 0;
}

// --------------------------- layer 1 (attention) ---------------------------
extern __shared__ char dynlds[];

__global__ void __launch_bounds__(1024)
k_layer1(const float* __restrict__ inputs, const int* __restrict__ text,
         const float* __restrict__ tmask,
         const float* __restrict__ h0, const float* __restrict__ c0,
         const float* __restrict__ w0, const float* __restrict__ k0v,
         const float* __restrict__ b1, const float* __restrict__ bw,
         const unsigned short* __restrict__ W1cat, const unsigned short* __restrict__ Wwp,
         unsigned short* __restrict__ HCAT, unsigned short* __restrict__ WV,
         float* __restrict__ c1, float* __restrict__ kappa,
         float* __restrict__ zbuf) {
    unsigned short* A = (unsigned short*)dynlds;              // [128][480] bf16
    float* fb   = (float*)(dynlds + (size_t)B_ * KL1 * 2);
    float* phi  = fb;                                         // 128*64
    float* al   = fb + 8192;                                  // 128*10
    float* be   = fb + 9472;
    float* ka   = fb + 10752;
    float* mixz = fb + 12032;                                 // 128*32

    const int tid = threadIdx.x, wave = tid >> 5, lane = tid & 31;
    const int half = lane >> 4, q = lane & 15;

    // ---- state init: h/w into LDS A, c/kappa into global state buffers ----
    for (int i = tid; i < B_ * H_; i += 1024) {
        const int b = i / H_, j = i % H_;
        A[b * KL1 + j] = f2bf(h0[i]);
        c1[i] = c0[i];
    }
    for (int i = tid; i < B_ * V_; i += 1024)
        A[(i / V_) * KL1 + 400 + (i % V_)] = f2bf(w0[i]);
    for (int i = tid; i < B_ * KK_; i += 1024) kappa[i] = k0v[i];
    __syncthreads();

    for (int t = 0; t < T_; ++t) {
        // x slot (cols 477..479)
        for (int i = tid; i < B_ * 3; i += 1024) {
            const int b = i / 3, c = i % 3;
            A[b * KL1 + 477 + c] = f2bf(inputs[((size_t)b * T_ + t) * 3 + c]);
        }
        __syncthreads();

        // gates: z = [h,w,x] @ W1cat^T + b1  (K = 480), A in LDS
        gates_gemm(A, KL1, W1cat, KL1, KL1 / 32, b1, zbuf, wave, lane);
        __threadfence(); __syncthreads();

        // LSTM cell (gate order i,f,g,o)
        for (int i = tid; i < B_ * H_; i += 1024) {
            const int b = i / H_, j = i % H_;
            const float* z = zbuf + (size_t)b * G_;
            const float ig = sigm(z[j]),        fg = sigm(z[400 + j]);
            const float gg = tanhf(z[800 + j]), og = sigm(z[1200 + j]);
            const float c = fg * c1[i] + ig * gg;
            c1[i] = c;
            const unsigned short hb = f2bf(og * tanhf(c));
            A[b * KL1 + j] = hb;                              // recurrent state
            HCAT[((size_t)t * B_ + b) * KOUT + j] = hb;
        }
        __threadfence(); __syncthreads();

        // mix = h1 @ Ww^T + bw : 8x2 tiles, waves 0..15, K = 416 (cols 400..415
        // of A hit zero columns of Wwp, so stale w values contribute nothing)
        if (wave < 16) {
            const int mt = wave & 7, nt = wave >> 3;
            const int m0 = mt * 16, n0 = nt * 16;
            v8f acc = {};
            acc = wmma_tile_k(A + (size_t)(m0 + q) * KL1,
                              Wwp + (size_t)(n0 + q) * KATT, KATT / 32, half, acc);
            const int n = n0 + q;
            const float bb = (n < 3 * KK_) ? bw[n] : 0.0f;
#pragma unroll
            for (int i = 0; i < 8; ++i)
                mixz[(m0 + i + 8 * half) * 32 + n] = acc[i] + bb;
        }
        __syncthreads();

        // kappa += exp(dk); alpha/beta = exp(.)
        for (int i = tid; i < B_ * KK_; i += 1024) {
            const int b = i / KK_, kk = i % KK_;
            const float kap = kappa[i] + __expf(mixz[b * 32 + 20 + kk]);
            kappa[i] = kap;
            al[i] = __expf(mixz[b * 32 + kk]);
            be[i] = __expf(mixz[b * 32 + 10 + kk]);
            ka[i] = kap;
        }
        __syncthreads();

        // phi[b,u] = sum_k alpha*exp(-beta*(kappa-u)^2) * mask
        for (int i = tid; i < B_ * U_; i += 1024) {
            const int b = i / U_, u = i % U_;
            float s = 0.0f;
#pragma unroll
            for (int kk = 0; kk < KK_; ++kk) {
                const float d = ka[b * KK_ + kk] - (float)u;
                s += al[b * KK_ + kk] * __expf(-be[b * KK_ + kk] * d * d);
            }
            phi[i] = s * tmask[i];
        }
        __syncthreads();

        // w[b,v] = sum_{u: text[b,u]==v} phi[b,u]; WV rows padded to 80
        for (int i = tid; i < B_ * WVS; i += 1024) {
            const int b = i / WVS, v = i % WVS;
            float s = 0.0f;
            if (v < V_)
                for (int u = 0; u < U_; ++u)
                    if (text[b * U_ + u] == v) s += phi[b * U_ + u];
            const unsigned short wb = f2bf(s);
            WV[((size_t)t * B_ + b) * WVS + v] = wb;
            if (v < V_) A[b * KL1 + 400 + v] = wb;            // input for t+1
        }
        __threadfence(); __syncthreads();
    }
}

// --------------------------- layers 2 & 3 ----------------------------------
__global__ void __launch_bounds__(1024)
k_layer23(const float* __restrict__ inputs,
          const float* __restrict__ h0L, const float* __restrict__ c0L,
          const float* __restrict__ bias, const unsigned short* __restrict__ Wcat,
          const unsigned short* __restrict__ WV,
          unsigned short* __restrict__ HCAT,
          float* __restrict__ cst, float* __restrict__ zbuf,
          int hsrc_col, int hdst_col) {
    unsigned short* A = (unsigned short*)dynlds;              // [128][896] bf16
    const int tid = threadIdx.x, wave = tid >> 5, lane = tid & 31;

    // ---- state init ----
    for (int i = tid; i < B_ * H_; i += 1024) {
        const int b = i / H_, j = i % H_;
        A[b * KL23 + 480 + j] = f2bf(h0L[i]);
        cst[i] = c0L[i];
    }
    for (int i = tid; i < B_ * 16; i += 1024)
        A[(i / 16) * KL23 + 880 + (i % 16)] = 0;
    __syncthreads();

    for (int t = 0; t < T_; ++t) {
        // async-stage h_prev (HCAT cols hsrc..hsrc+399) and wv into LDS A
        for (int i = tid; i < 128 * 50; i += 1024) {
            const int r = i / 50, c = i % 50;
            cp16_g2l(HCAT + ((size_t)t * B_ + r) * KOUT + hsrc_col + c * 8,
                     A + r * KL23 + c * 8);
        }
        for (int i = tid; i < 128 * 10; i += 1024) {
            const int r = i / 10, c = i % 10;
            cp16_g2l(WV + ((size_t)t * B_ + r) * WVS + c * 8,
                     A + r * KL23 + 400 + c * 8);
        }
        async_wait0();
        __syncthreads();
        // x slot (cols 477..479; overwrites WV-row padding)
        for (int i = tid; i < B_ * 3; i += 1024) {
            const int b = i / 3, c = i % 3;
            A[b * KL23 + 477 + c] = f2bf(inputs[((size_t)b * T_ + t) * 3 + c]);
        }
        __syncthreads();

        gates_gemm(A, KL23, Wcat, KL23, KL23 / 32, bias, zbuf, wave, lane);
        __threadfence(); __syncthreads();

        for (int i = tid; i < B_ * H_; i += 1024) {
            const int b = i / H_, j = i % H_;
            const float* z = zbuf + (size_t)b * G_;
            const float ig = sigm(z[j]),        fg = sigm(z[400 + j]);
            const float gg = tanhf(z[800 + j]), og = sigm(z[1200 + j]);
            const float c = fg * cst[i] + ig * gg;
            cst[i] = c;
            const unsigned short hb = f2bf(og * tanhf(c));
            A[b * KL23 + 480 + j] = hb;                       // recurrent state
            HCAT[((size_t)t * B_ + b) * KOUT + hdst_col + j] = hb;
        }
        __threadfence(); __syncthreads();
    }
}

// --------------------------- output projection ------------------------------
// y[b,t,n] = HCAT[t*128+b, :] @ Wop^T + bo.  Block bid owns rows [bid*128,+128):
// A chunks async-staged to LDS, each wave = one 16-row tile x 8 n-tiles.
__global__ void __launch_bounds__(256)
k_out(const unsigned short* __restrict__ HCAT, const unsigned short* __restrict__ Wop,
      const float* __restrict__ bo, float* __restrict__ out) {
    __shared__ unsigned short As[128 * KCH];                  // 16 KB
    const int tid = threadIdx.x, wave = tid >> 5, lane = tid & 31;
    const int half = lane >> 4, q = lane & 15;
    const size_t mbase = (size_t)blockIdx.x * 128;

    v8f acc[8];
#pragma unroll
    for (int nt = 0; nt < 8; ++nt) acc[nt] = (v8f){};

    for (int kc = 0; kc < KOUT / KCH; ++kc) {                 // 19 chunks
        __syncthreads();                                      // protect As reuse
        for (int i = tid; i < 128 * (KCH / 8); i += 256) {    // 1024 x 16B
            const int r = i >> 3, c = i & 7;
            cp16_g2l(HCAT + (mbase + r) * KOUT + kc * KCH + c * 8,
                     As + r * KCH + c * 8);
        }
        async_wait0();
        __syncthreads();
        const unsigned short* asrow = As + (wave * 16 + q) * KCH;
        for (int ks = 0; ks < KCH / 32; ++ks) {
            const int k0l = ks * 32;
            v8u au;
#pragma unroll
            for (int v = 0; v < 8; ++v)
                au[v] = *(const unsigned int*)(asrow + k0l +
                            2 * v + 16 * half);
            const v16bf a = __builtin_bit_cast(v16bf, au);
            const int k0g = kc * KCH + k0l;
#pragma unroll
            for (int nt = 0; nt < 8; ++nt) {
                const unsigned short* brow = Wop + (size_t)(nt * 16 + q) * KOUT;
                v8u bu;
#pragma unroll
                for (int v = 0; v < 8; ++v)
                    bu[v] = *(const unsigned int*)(brow + k0g + 2 * v + 16 * half);
                acc[nt] = __builtin_amdgcn_wmma_f32_16x16x32_bf16(
                              false, a, false, __builtin_bit_cast(v16bf, bu),
                              (short)0, acc[nt], false, false);
            }
        }
    }
#pragma unroll
    for (int nt = 0; nt < 8; ++nt) {
        const int n = nt * 16 + q;
        if (n < O_) {
            const float bb = bo[n];
#pragma unroll
            for (int i = 0; i < 8; ++i) {
                const int m = (int)mbase + wave * 16 + i + 8 * half;  // m = t*128+b
                const int tt = m / B_, b = m % B_;
                out[((size_t)b * T_ + tt) * O_ + n] = acc[nt][i] + bb;
            }
        }
    }
}

// --------------------------- launcher ---------------------------------------
extern "C" void kernel_launch(void* const* d_in, const int* in_sizes, int n_in,
                              void* d_out, int out_size, void* d_ws, size_t ws_size,
                              hipStream_t stream) {
    const float* inputs = (const float*)d_in[0];
    const int*   text   = (const int*)  d_in[1];
    const float* tmask  = (const float*)d_in[2];
    const float* h0     = (const float*)d_in[3];
    const float* c0     = (const float*)d_in[4];
    const float* w0     = (const float*)d_in[5];
    const float* k0v    = (const float*)d_in[6];
    const float* Wih1 = (const float*)d_in[7],  *Whh1 = (const float*)d_in[8],  *b1 = (const float*)d_in[9];
    const float* Wih2 = (const float*)d_in[10], *Whh2 = (const float*)d_in[11], *b2 = (const float*)d_in[12];
    const float* Wih3 = (const float*)d_in[13], *Whh3 = (const float*)d_in[14], *b3 = (const float*)d_in[15];
    const float* Ww = (const float*)d_in[16], *bw = (const float*)d_in[17];
    const float* Wo = (const float*)d_in[18], *bo = (const float*)d_in[19];
    float* out = (float*)d_out;
    (void)in_sizes; (void)n_in; (void)out_size; (void)ws_size;

    size_t off = 0;
    auto carve = [&](size_t bytes) -> char* {
        char* p = (char*)d_ws + off;
        off = (off + bytes + 255) & ~(size_t)255;
        return p;
    };
    unsigned short* W1cat = (unsigned short*)carve((size_t)G_ * KL1 * 2);
    unsigned short* W2cat = (unsigned short*)carve((size_t)G_ * KL23 * 2);
    unsigned short* W3cat = (unsigned short*)carve((size_t)G_ * KL23 * 2);
    unsigned short* Wwp   = (unsigned short*)carve((size_t)32 * KATT * 2);
    unsigned short* Wop   = (unsigned short*)carve((size_t)128 * KOUT * 2);
    unsigned short* HCAT  = (unsigned short*)carve((size_t)T_ * B_ * KOUT * 2);
    unsigned short* WV    = (unsigned short*)carve((size_t)T_ * B_ * WVS * 2 + 256);
    float* zbuf  = (float*)carve((size_t)B_ * G_ * 4);
    float* c1    = (float*)carve((size_t)B_ * H_ * 4);
    float* c2    = (float*)carve((size_t)B_ * H_ * 4);
    float* c3    = (float*)carve((size_t)B_ * H_ * 4);
    float* kappa = (float*)carve((size_t)B_ * KK_ * 4);

    // pack weights (bf16, K-permuted to match LDS A layouts)
    k_cvt1 <<<512, 256, 0, stream>>>(Wih1, Whh1, W1cat);
    k_cvt23<<<512, 256, 0, stream>>>(Wih2, Whh2, W2cat);
    k_cvt23<<<512, 256, 0, stream>>>(Wih3, Whh3, W3cat);
    k_cvt_pad<<<64, 256, 0, stream>>>(Ww, Wwp, 32, KATT, 3 * KK_, H_);
    k_cvt_pad<<<256, 256, 0, stream>>>(Wo, Wop, 128, KOUT, O_, 1200);
    k_pad_hcat<<<1024, 256, 0, stream>>>(HCAT);

    const size_t lds1  = (size_t)B_ * KL1 * 2 + 16128 * 4;    // 187,392 B
    const size_t lds23 = (size_t)B_ * KL23 * 2;               // 229,376 B

    k_layer1<<<1, 1024, lds1, stream>>>(inputs, text, tmask, h0, c0, w0, k0v,
                                        b1, bw, W1cat, Wwp, HCAT, WV,
                                        c1, kappa, zbuf);
    k_layer23<<<1, 1024, lds23, stream>>>(inputs, h0 + B_ * H_, c0 + B_ * H_,
                                          b2, W2cat, WV, HCAT, c2, zbuf,
                                          /*hsrc*/0,   /*hdst*/400);
    k_layer23<<<1, 1024, lds23, stream>>>(inputs, h0 + 2 * B_ * H_, c0 + 2 * B_ * H_,
                                          b3, W3cat, WV, HCAT, c3, zbuf,
                                          /*hsrc*/400, /*hdst*/800);

    k_out<<<600, 256, 0, stream>>>(HCAT, Wop, bo, out);
}
